// GraphLayer_1700807049908
// MI455X (gfx1250) — matrix operand order repaired and checked by reading, not compiled, not run on hardware
//
#include <hip/hip_runtime.h>

// ---------------------------------------------------------------------------
// MI455X (gfx1250) fused graph-attention layer.
// All GEMMs use v_wmma_f32_16x16x32_bf16 (fp32 -> bf16 operands, f32 accum).
// GEMM: 64x64 macro-tile per 128-thread workgroup; B staged in LDS in
//       fragment order (contiguous 32B per lane -> ds_load_b128), with
//       global_prefetch_b8 of the next sub-tile. Attention core fused
//       per-batch with a 64KB LDS score tile and register-cached fragments.
// ---------------------------------------------------------------------------

typedef __attribute__((ext_vector_type(16))) __bf16 v16bf;
typedef __attribute__((ext_vector_type(8)))  float  v8f;

// ----- WMMA fragment builders (contiguous k order K = kh*16 + j, same for A and B) ----------------
// A (16-bit, 16x32, row-major source): lane = row (lane&15), kh = lane>>4.
//   element j    -> K = kk + kh*16 + j
__device__ __forceinline__ v16bf frag_a_rm(const float* __restrict__ row,
                                           int kk, int kh) {
  v16bf f;
  const float* p = row + kk + kh * 16;
#pragma unroll
  for (int j = 0; j < 8; ++j) f[j] = (__bf16)p[j];
#pragma unroll
  for (int j = 0; j < 8; ++j) f[8 + j] = (__bf16)p[8 + j];
  return f;
}

// B (16-bit, 32x16, row-major [K x N] source): lane = col, K = kk + kh*16 + j
__device__ __forceinline__ v16bf frag_b_rm(const float* __restrict__ B, int ldb,
                                           int n, int kk, int kh) {
  v16bf f;
  const float* p = B + (size_t)(kk + kh * 16) * ldb + n;
#pragma unroll
  for (int j = 0; j < 16; ++j) f[j] = (__bf16)p[(size_t)j * ldb];
  return f;
}

// B from a transposed source (element (k,n) stored at rowT[n*ldt + k]):
// contiguous in k -> coalesced loads. Used for Q @ K^T.
__device__ __forceinline__ v16bf frag_b_tr(const float* __restrict__ rowT,
                                           int kk, int kh) {
  v16bf f;
  const float* p = rowT + kk + kh * 16;
#pragma unroll
  for (int j = 0; j < 16; ++j) f[j] = (__bf16)p[j];
  return f;
}

// ----- Macro-tiled GEMM:  C[M,N] = A[M,K] @ B[K,N] + bias[N] ----------------
// grid = (N/64, M/64), 128 threads (4 waves). Wave w owns rows
// [m0+16w, m0+16w+16) and all four 16-wide n-tiles. Per 32-deep k-step the
// 32x64 B sub-tile is staged into LDS in *fragment order*:
//   BsT[col * 40 + k]  (bf16, 40-element column stride)
// so a lane's 16 B values (K = kh*16+j at fixed col) are 32 contiguous,
// 16B-aligned bytes -> two ds_load_b128, bank-conflict-free.
__global__ __launch_bounds__(128) void gemm_bias64(
    const float* __restrict__ A, const float* __restrict__ B,
    const float* __restrict__ bias, float* __restrict__ C, int K, int N) {
  __shared__ alignas(16) __bf16 BsT[64 * 40];  // 5 KB

  const int tid  = threadIdx.x;
  const int wave = tid >> 5;
  const int lane = tid & 31;
  const int rr = lane & 15;      // row (A) / col (B) within 16x16 tile
  const int kh = lane >> 4;      // half-wave selector
  const int m0 = blockIdx.y * 64;
  const int n0 = blockIdx.x * 64;

  const float* arow = A + (size_t)(m0 + wave * 16 + rr) * K;

  const v8f vzero = {0.f, 0.f, 0.f, 0.f, 0.f, 0.f, 0.f, 0.f};
  v8f acc[4];
#pragma unroll
  for (int t = 0; t < 4; ++t) acc[t] = vzero;

  // Staging map: thread owns k-row pair (2rp, 2rp+1) x 8 columns, so the two
  // bf16 stores per column are adjacent -> merge into ds_store_b32.
  const int rp = tid >> 3;          // 0..15
  const int cb = (tid & 7) * 8;     // 0,8,...,56

  for (int kk = 0; kk < K; kk += 32) {
    __syncthreads();  // WAR: previous iteration's fragment reads done
    {
      const float* s0 = B + (size_t)(kk + 2 * rp) * N + n0 + cb;
      const float* s1 = s0 + N;
#pragma unroll
      for (int j = 0; j < 8; ++j) {
        BsT[(cb + j) * 40 + 2 * rp]     = (__bf16)s0[j];
        BsT[(cb + j) * 40 + 2 * rp + 1] = (__bf16)s1[j];
      }
      if (kk + 32 < K)  // pull next sub-tile toward the WGP (global_prefetch)
        __builtin_prefetch(s0 + (size_t)32 * N, 0, 0);
    }
    __syncthreads();  // RAW: sub-tile visible to all waves

    const v16bf a = frag_a_rm(arow, kk, kh);  // reused across 4 n-tiles
#pragma unroll
    for (int t = 0; t < 4; ++t) {
      v16bf bfrag;
      const __bf16* p = &BsT[(t * 16 + rr) * 40 + kh * 16];
#pragma unroll
      for (int j = 0; j < 16; ++j) bfrag[j] = p[j];  // 2x ds_load_b128
      acc[t] = __builtin_amdgcn_wmma_f32_16x16x32_bf16(
          false, a, false, bfrag, (short)0, acc[t], false, false);
    }
  }

  const int mb = m0 + wave * 16 + kh * 8;  // C/D layout: VGPR j -> M = mb + j
#pragma unroll
  for (int t = 0; t < 4; ++t) {
    const int n = n0 + t * 16 + rr;
    const float bv = bias[n];
#pragma unroll
    for (int j = 0; j < 8; ++j) C[(size_t)(mb + j) * N + n] = acc[t][j] + bv;
  }
}

// ----- Fused attention core: one workgroup (8 waves) per batch --------------
// Phase 1: S = (Q @ K^T) / 16            (WMMA -> LDS, A-frags register-cached)
// Phase 2: S *= sigmoid(edgeMLP); softmax rows (lane-pair shuffle reductions)
// Phase 3: out = P @ V                   (WMMA, V B-frags register-cached)
__global__ __launch_bounds__(256) void attn_kernel(
    const float* __restrict__ qg, const float* __restrict__ kg,
    const float* __restrict__ vg, const float* __restrict__ area,
    const float* __restrict__ co, const float* __restrict__ We1,
    const float* __restrict__ be1, const float* __restrict__ We2,
    const float* __restrict__ be2, float* __restrict__ out_raw) {
  __shared__ float Sf[128 * 128];  // 64 KB score/prob tile

  const int b = blockIdx.x;
  const int wave = threadIdx.x >> 5;
  const int lane = threadIdx.x & 31;
  const int rr = lane & 15;
  const int kh = lane >> 4;

  const float* qb = qg + (size_t)b * 128 * 256;
  const float* kb = kg + (size_t)b * 128 * 256;
  const float* vb = vg + (size_t)b * 128 * 256;

  // ---- Phase 1: wave w owns rows [16w,16w+16); 8 column tiles -------------
  {
    const float* qrow = qb + (size_t)(wave * 16 + rr) * 256;
    v16bf afr[8];  // the wave's whole Q row-block, loaded once
#pragma unroll
    for (int kt = 0; kt < 8; ++kt) afr[kt] = frag_a_rm(qrow, kt * 32, kh);

    for (int nt = 0; nt < 8; ++nt) {
      v8f acc = {0.f, 0.f, 0.f, 0.f, 0.f, 0.f, 0.f, 0.f};
      const float* krow = kb + (size_t)(nt * 16 + rr) * 256;  // K^T contiguous
#pragma unroll
      for (int kt = 0; kt < 8; ++kt) {
        v16bf bmat = frag_b_tr(krow, kt * 32, kh);
        acc = __builtin_amdgcn_wmma_f32_16x16x32_bf16(
            false, afr[kt], false, bmat, (short)0, acc, false, false);
      }
      const int n = nt * 16 + rr;
      const int mb = wave * 16 + kh * 8;
#pragma unroll
      for (int j = 0; j < 8; ++j)
        Sf[(mb + j) * 128 + n] = acc[j] * 0.0625f;  // 1/sqrt(256)
    }
  }
  __syncthreads();

  // ---- Phase 2: edge gating + softmax. 2 threads per row (64 cols each) ---
  {
    const int row = threadIdx.x >> 1;
    const int half = threadIdx.x & 1;
    const int c0 = half * 64;
    const float a = area[b * 128 + row];
    const float* corow = co + row * 128;
    const float bias2 = be2[0];

    float mx = -3.402823e38f;
    for (int m = c0; m < c0 + 64; ++m) {
      const float c = corow[m];
      float acc = bias2;
      for (int h = 0; h < 256; ++h) {
        float e = fmaf(a, We1[h], fmaf(c, We1[256 + h], be1[h]));
        acc = fmaf(fmaxf(e, 0.0f), We2[h], acc);
      }
      const float ew = 1.0f / (1.0f + __expf(-acc));  // sigmoid
      const float s = Sf[row * 128 + m] * ew;
      Sf[row * 128 + m] = s;
      mx = fmaxf(mx, s);
    }
    mx = fmaxf(mx, __shfl_xor(mx, 1));  // combine with partner lane
    float sum = 0.0f;
    for (int m = c0; m < c0 + 64; ++m) {
      const float e = __expf(Sf[row * 128 + m] - mx);
      Sf[row * 128 + m] = e;
      sum += e;
    }
    sum += __shfl_xor(sum, 1);
    const float inv = 1.0f / sum;
    for (int m = c0; m < c0 + 64; ++m) Sf[row * 128 + m] *= inv;
  }
  __syncthreads();

  // ---- Phase 3: out = P @ V. Wave w owns n-tiles {2w, 2w+1}: its 8 strided
  // V B-fragments are loaded once into registers, then reused across all 8
  // m-tiles, whose A-fragments come from LDS probs (contiguous 32B runs).
  {
    float* outb = out_raw + (size_t)b * 128 * 256;
    v16bf vfr[2][4];
#pragma unroll
    for (int tn = 0; tn < 2; ++tn)
#pragma unroll
      for (int kt = 0; kt < 4; ++kt)
        vfr[tn][kt] =
            frag_b_rm(vb, 256, (wave * 2 + tn) * 16 + rr, kt * 32, kh);

    for (int mt = 0; mt < 8; ++mt) {
      v8f a0 = {0.f, 0.f, 0.f, 0.f, 0.f, 0.f, 0.f, 0.f};
      v8f a1 = {0.f, 0.f, 0.f, 0.f, 0.f, 0.f, 0.f, 0.f};
#pragma unroll
      for (int kt = 0; kt < 4; ++kt) {
        v16bf a;
        const int base = (mt * 16 + rr) * 128 + kt * 32 + kh * 16;
#pragma unroll
        for (int j = 0; j < 8; ++j) a[j] = (__bf16)Sf[base + j];
#pragma unroll
        for (int j = 0; j < 8; ++j) a[8 + j] = (__bf16)Sf[base + 8 + j];
        a0 = __builtin_amdgcn_wmma_f32_16x16x32_bf16(
            false, a, false, vfr[0][kt], (short)0, a0, false, false);
        a1 = __builtin_amdgcn_wmma_f32_16x16x32_bf16(
            false, a, false, vfr[1][kt], (short)0, a1, false, false);
      }
      const int mb = mt * 16 + kh * 8;
      const int n0 = (wave * 2) * 16 + rr;
      const int n1 = (wave * 2 + 1) * 16 + rr;
#pragma unroll
      for (int j = 0; j < 8; ++j) {
        outb[(size_t)(mb + j) * 256 + n0] = a0[j];
        outb[(size_t)(mb + j) * 256 + n1] = a1[j];
      }
    }
  }
}

// ----- Row LayerNorm over H=256: one wave per row, 8 values per lane --------
__device__ __forceinline__ float wave_allreduce_add(float v) {
#pragma unroll
  for (int off = 16; off; off >>= 1) v += __shfl_xor(v, off);
  return v;
}

__global__ __launch_bounds__(256) void ln256_kernel(
    const float* __restrict__ src, float* __restrict__ dst,
    const float* __restrict__ g, const float* __restrict__ be) {
  const int row = blockIdx.x * 8 + (threadIdx.x >> 5);
  const int lane = threadIdx.x & 31;
  const float* s = src + (size_t)row * 256;

  float v[8];
  float mean = 0.f;
#pragma unroll
  for (int i = 0; i < 8; ++i) { v[i] = s[lane * 8 + i]; mean += v[i]; }
  mean = wave_allreduce_add(mean) * (1.0f / 256.0f);
  float var = 0.f;
#pragma unroll
  for (int i = 0; i < 8; ++i) { const float d = v[i] - mean; var += d * d; }
  var = wave_allreduce_add(var) * (1.0f / 256.0f);
  const float inv = rsqrtf(var + 1e-5f);
#pragma unroll
  for (int i = 0; i < 8; ++i) {
    const int c = lane * 8 + i;
    dst[(size_t)row * 256 + c] = (v[i] - mean) * inv * g[c] + be[c];
  }
}

// ----- Final: y = LN( relu(LN(t, g_out, b_out)) + x_proj, g2, b2 ) ----------
__global__ __launch_bounds__(256) void final_kernel(
    const float* __restrict__ t, const float* __restrict__ xp,
    const float* __restrict__ g_out, const float* __restrict__ b_out,
    const float* __restrict__ g2, const float* __restrict__ b2,
    float* __restrict__ y) {
  const int row = blockIdx.x * 8 + (threadIdx.x >> 5);
  const int lane = threadIdx.x & 31;
  const float* s = t + (size_t)row * 256;
  const float* xr = xp + (size_t)row * 256;

  float v[8];
  float mean = 0.f;
#pragma unroll
  for (int i = 0; i < 8; ++i) { v[i] = s[lane * 8 + i]; mean += v[i]; }
  mean = wave_allreduce_add(mean) * (1.0f / 256.0f);
  float var = 0.f;
#pragma unroll
  for (int i = 0; i < 8; ++i) { const float d = v[i] - mean; var += d * d; }
  var = wave_allreduce_add(var) * (1.0f / 256.0f);
  float inv = rsqrtf(var + 1e-5f);

  // relu(LN1) + residual
  float mean2 = 0.f;
#pragma unroll
  for (int i = 0; i < 8; ++i) {
    const int c = lane * 8 + i;
    float u = (v[i] - mean) * inv * g_out[c] + b_out[c];
    u = fmaxf(u, 0.0f) + xr[c];
    v[i] = u;
    mean2 += u;
  }
  mean2 = wave_allreduce_add(mean2) * (1.0f / 256.0f);
  float var2 = 0.f;
#pragma unroll
  for (int i = 0; i < 8; ++i) { const float d = v[i] - mean2; var2 += d * d; }
  var2 = wave_allreduce_add(var2) * (1.0f / 256.0f);
  inv = rsqrtf(var2 + 1e-5f);
#pragma unroll
  for (int i = 0; i < 8; ++i) {
    const int c = lane * 8 + i;
    y[(size_t)row * 256 + c] = (v[i] - mean2) * inv * g2[c] + b2[c];
  }
}

// ---------------------------------------------------------------------------
extern "C" void kernel_launch(void* const* d_in, const int* in_sizes, int n_in,
                              void* d_out, int out_size, void* d_ws,
                              size_t ws_size, hipStream_t stream) {
  const float* x     = (const float*)d_in[0];   // [32,128,512]
  const float* area  = (const float*)d_in[1];   // [32,128]
  const float* co    = (const float*)d_in[2];   // [128,128]
  const float* Wf    = (const float*)d_in[3];   // [512,256]
  const float* bf    = (const float*)d_in[4];
  const float* Wq    = (const float*)d_in[5];
  const float* bq    = (const float*)d_in[6];
  const float* Wk    = (const float*)d_in[7];
  const float* bk    = (const float*)d_in[8];
  const float* Wv    = (const float*)d_in[9];
  const float* bv    = (const float*)d_in[10];
  const float* We1   = (const float*)d_in[11];  // [2,256]
  const float* be1   = (const float*)d_in[12];
  const float* We2   = (const float*)d_in[13];  // [256,1]
  const float* be2   = (const float*)d_in[14];
  const float* Wo    = (const float*)d_in[15];
  const float* bo    = (const float*)d_in[16];
  const float* g_out = (const float*)d_in[17];
  const float* b_out = (const float*)d_in[18];
  const float* g1    = (const float*)d_in[19];
  const float* b1    = (const float*)d_in[20];
  const float* g2    = (const float*)d_in[21];
  const float* b2    = (const float*)d_in[22];

  float* ws = (float*)d_ws;
  const size_t M1 = (size_t)4096 * 256;  // 1M floats = 4 MB
  float* xp = ws;            // x_proj   [4096,256]   (live to the end)
  float* qb = ws + 1 * M1;   // q
  float* kb = ws + 2 * M1;   // k
  float* vb = ws + 3 * M1;   // v
  float* ob = ws + 4 * M1;   // attn out; LN1 applied in-place
  float* tb = ws + 1 * M1;   // t = ln1 @ Wo + bo  (reuses q, dead by then)

  const dim3 blk128(128);
  const dim3 gTiles(4, 64);  // N/64 x M/64 for M=4096, N=256

  // 1) x_proj = x @ Wf + bf   (K=512)
  gemm_bias64<<<gTiles, blk128, 0, stream>>>(x, Wf, bf, xp, 512, 256);
  // 2-4) q,k,v = x_proj @ W* + b*   (K=256)
  gemm_bias64<<<gTiles, blk128, 0, stream>>>(xp, Wq, bq, qb, 256, 256);
  gemm_bias64<<<gTiles, blk128, 0, stream>>>(xp, Wk, bk, kb, 256, 256);
  gemm_bias64<<<gTiles, blk128, 0, stream>>>(xp, Wv, bv, vb, 256, 256);
  // 5) fused attention core, one workgroup per batch
  attn_kernel<<<32, 256, 0, stream>>>(qb, kb, vb, area, co, We1, be1, We2,
                                      be2, ob);
  // 6) LN1 in-place
  ln256_kernel<<<512, 256, 0, stream>>>(ob, ob, g1, b1);
  // 7) t = ln1 @ Wo + bo
  gemm_bias64<<<gTiles, blk128, 0, stream>>>(ob, Wo, bo, tb, 256, 256);
  // 8) y = LN(relu(LN(t)) + x_proj)
  final_kernel<<<512, 256, 0, stream>>>(tb, xp, g_out, b_out, g2, b2,
                                        (float*)d_out);
}
